// BitLinear_81046032875543
// MI455X (gfx1250) — matrix operand order repaired and checked by reading, not compiled, not run on hardware
//
#include <hip/hip_runtime.h>

typedef __attribute__((ext_vector_type(16))) _Float16 v16h;
typedef __attribute__((ext_vector_type(8)))  _Float16 v8h;
typedef __attribute__((ext_vector_type(4)))  _Float16 v4h;
typedef __attribute__((ext_vector_type(8)))  float    v8f;
typedef __attribute__((ext_vector_type(4)))  int      v4i;

#define M_DIM 8192      // 4 * 2048
#define K_DIM 4096
#define N_DIM 11008
#define THRESH_FACTOR 0.7f
#define RED_BLOCKS 1024

// GEMM tiling
#define BT_M 256
#define BT_N 256
#define KSTEP 32
#define NSTAGE (K_DIM / KSTEP)          // 128
#define ROW_BYTES 64                    // KSTEP * 2
#define ROW_STRIDE 80                   // padded: 16 lanes -> 16 distinct LDS banks
#define A_TILE_BYTES (BT_M * ROW_STRIDE)      // 20480
#define STAGE_BYTES  (2 * A_TILE_BYTES)       // 40960 (A then B)

#define HAS_ASYNC_CP __has_builtin(__builtin_amdgcn_global_load_async_to_lds_b128)

// builtin signature (from hipcc diagnostic): arg0 = AS1 v4i* (non-const),
// arg1 = LDS side. uintptr_t round-trip: generic->AS1 is identity; low 32 bits
// of a generic LDS address are the LDS byte offset (ISA Ch10 aperture rules).
#define AS1P(p) ((__attribute__((address_space(1))) v4i*)(uintptr_t)(p))
#define AS3P(p) ((__attribute__((address_space(3))) v4i*)(uintptr_t)(p))

// ---------------------------------------------------------------------------
// Stage 1a: per-block partial sums of |W| (fixed order -> deterministic)
// ---------------------------------------------------------------------------
__global__ void k_abs_partial(const float* __restrict__ w, float* __restrict__ partial, int n4) {
    __shared__ float sm[256];
    const float4* w4 = (const float4*)w;
    float s = 0.0f;
    for (int i = blockIdx.x * blockDim.x + threadIdx.x; i < n4; i += gridDim.x * blockDim.x) {
        float4 v = w4[i];
        s += fabsf(v.x) + fabsf(v.y) + fabsf(v.z) + fabsf(v.w);
    }
    sm[threadIdx.x] = s;
    __syncthreads();
    for (int off = 128; off > 0; off >>= 1) {
        if (threadIdx.x < off) sm[threadIdx.x] += sm[threadIdx.x + off];
        __syncthreads();
    }
    if (threadIdx.x == 0) partial[blockIdx.x] = sm[0];
}

// ---------------------------------------------------------------------------
// Stage 1b: final reduce -> threshold = mean(|W|) * 0.7
// ---------------------------------------------------------------------------
__global__ void k_thresh(const float* __restrict__ partial, float* __restrict__ thresh,
                         int nparts, float inv_n) {
    __shared__ float sm[256];
    float s = 0.0f;
    for (int i = threadIdx.x; i < nparts; i += 256) s += partial[i];
    sm[threadIdx.x] = s;
    __syncthreads();
    for (int off = 128; off > 0; off >>= 1) {
        if (threadIdx.x < off) sm[threadIdx.x] += sm[threadIdx.x + off];
        __syncthreads();
    }
    if (threadIdx.x == 0) *thresh = sm[0] * inv_n * THRESH_FACTOR;
}

// ---------------------------------------------------------------------------
// Stage 2: ternarize + scale weight into f16  wq[o,k] = sign(w)*(|w|>t)*scale[o]
// ---------------------------------------------------------------------------
__global__ void k_quant(const float* __restrict__ w, const float* __restrict__ scale,
                        const float* __restrict__ thresh, _Float16* __restrict__ wq, int n4) {
    int i = blockIdx.x * blockDim.x + threadIdx.x;
    if (i >= n4) return;
    const float t = *thresh;
    float4 v = ((const float4*)w)[i];
    int o = (i * 4) >> 12;            // element / 4096 -> output row
    float sc = scale[o];
    v4h q;
    float c;
    c = v.x; q[0] = (_Float16)((fabsf(c) > t) ? (c > 0.0f ? sc : -sc) : 0.0f);
    c = v.y; q[1] = (_Float16)((fabsf(c) > t) ? (c > 0.0f ? sc : -sc) : 0.0f);
    c = v.z; q[2] = (_Float16)((fabsf(c) > t) ? (c > 0.0f ? sc : -sc) : 0.0f);
    c = v.w; q[3] = (_Float16)((fabsf(c) > t) ? (c > 0.0f ? sc : -sc) : 0.0f);
    ((v4h*)wq)[i] = q;
}

// ---------------------------------------------------------------------------
// Stage 3: x fp32 -> f16 (staged once; 67 MB stays L2-resident for the GEMM)
// ---------------------------------------------------------------------------
__global__ void k_xcvt(const float* __restrict__ x, _Float16* __restrict__ x16, int n4) {
    int i = blockIdx.x * blockDim.x + threadIdx.x;
    if (i >= n4) return;
    float4 v = ((const float4*)x)[i];
    v4h q;
    q[0] = (_Float16)v.x; q[1] = (_Float16)v.y; q[2] = (_Float16)v.z; q[3] = (_Float16)v.w;
    ((v4h*)x16)[i] = q;
}

// ---------------------------------------------------------------------------
// Copy one 64-byte row segment (global K-major) into LDS.
// Async path: 4 x global_load_async_to_lds_b128 (ASYNCcnt, bypasses VGPRs).
// Fallback: synchronous reg round-trip (DScnt, covered by __syncthreads).
// ---------------------------------------------------------------------------
__device__ inline void issue_row_copy(const _Float16* gsrc, char* ldst) {
#if HAS_ASYNC_CP
    __builtin_amdgcn_global_load_async_to_lds_b128(AS1P(gsrc), AS3P(ldst), 0,  0);
    __builtin_amdgcn_global_load_async_to_lds_b128(AS1P(gsrc), AS3P(ldst), 16, 0);
    __builtin_amdgcn_global_load_async_to_lds_b128(AS1P(gsrc), AS3P(ldst), 32, 0);
    __builtin_amdgcn_global_load_async_to_lds_b128(AS1P(gsrc), AS3P(ldst), 48, 0);
#else
    const v8h* g = (const v8h*)gsrc;
    v8h t0 = g[0], t1 = g[1], t2 = g[2], t3 = g[3];
    v8h* l = (v8h*)ldst;
    l[0] = t0; l[1] = t1; l[2] = t2; l[3] = t3;
#endif
}

__device__ inline void wait_async_keep4() {
#if HAS_ASYNC_CP
# if __has_builtin(__builtin_amdgcn_s_wait_asynccnt)
    __builtin_amdgcn_s_wait_asynccnt(4);
# else
    asm volatile("s_wait_asynccnt 0x4" ::: "memory");
# endif
#endif
}

__device__ inline void wait_async_all() {
#if HAS_ASYNC_CP
# if __has_builtin(__builtin_amdgcn_s_wait_asynccnt)
    __builtin_amdgcn_s_wait_asynccnt(0);
# else
    asm volatile("s_wait_asynccnt 0x0" ::: "memory");
# endif
#endif
}

__device__ inline v16h cat8(v8h lo, v8h hi) {
    return __builtin_shufflevector(lo, hi, 0, 1, 2, 3, 4, 5, 6, 7,
                                           8, 9, 10, 11, 12, 13, 14, 15);
}

// ---------------------------------------------------------------------------
// Stage 4: WMMA GEMM  y[m,n] = sum_k x16[m,k] * wq16[n,k]
//   512 threads = 16 waves (wave32); wave tile 64x64; block tile 256x256.
//   Double-buffered LDS staging via async copies; ds_load_b128 fragment reads.
//   A frag (16x32 f16): lane<16 -> M=lane, K={0..15}; lane>=16 ->
//   M=lane-16, K={16..31}. B frag (32x16 f16): N=lane%16,
//   K=16*(lane/16)+[0..15]. Both map to contiguous 16/32B chunks of K-major rows.
// ---------------------------------------------------------------------------
__global__ __launch_bounds__(512) void k_gemm(const _Float16* __restrict__ x16,
                                              const _Float16* __restrict__ wq16,
                                              float* __restrict__ out) {
    __shared__ alignas(16) char smem[2 * STAGE_BYTES];   // 80 KB of the 320 KB WGP LDS

    const int tid   = threadIdx.x;
    const int lane  = tid & 31;
    const int wave  = tid >> 5;          // 0..15
    const int wm    = wave >> 2;         // 0..3
    const int wn    = wave & 3;          // 0..3
    const int lmod  = lane & 15;
    const int lhalf = lane >> 4;         // 0 or 1

    const int m0 = blockIdx.y * BT_M;
    const int n0 = blockIdx.x * BT_N;

    // --- copy-role assignment: threads 0..255 own A rows, 256..511 own B rows
    const bool isA = tid < 256;
    const int  crow = isA ? tid : tid - 256;
    const _Float16* gsrc = isA ? (x16  + (size_t)(m0 + crow) * K_DIM)
                               : (wq16 + (size_t)(n0 + crow) * K_DIM);
    char* lrow0 = smem + (isA ? 0 : A_TILE_BYTES) + crow * ROW_STRIDE;

    v8f acc[4][4];
#pragma unroll
    for (int i = 0; i < 4; ++i)
#pragma unroll
        for (int j = 0; j < 4; ++j) acc[i][j] = (v8f){};

    // fragment base offsets within a stage (bytes)
    char* aBase = smem + (size_t)(wm * 64 + lmod) * ROW_STRIDE + lhalf * 32;
    char* bBase = smem + A_TILE_BYTES + (size_t)(wn * 64 + lmod) * ROW_STRIDE + lhalf * 32;

    // prologue: stage 0 copy
    issue_row_copy(gsrc, lrow0);

    for (int s = 0; s < NSTAGE; ++s) {
        if (s > 0) __syncthreads();                 // prior reads of buf[(s+1)&1] done
        if (s + 1 < NSTAGE) {
            issue_row_copy(gsrc + (s + 1) * KSTEP, lrow0 + ((s + 1) & 1) * STAGE_BYTES);
            wait_async_keep4();                     // stage s's 4 copies complete
        } else {
            wait_async_all();
        }
        __syncthreads();                            // stage s visible to all waves

        char* sa = aBase + (s & 1) * STAGE_BYTES;
        char* sb = bBase + (s & 1) * STAGE_BYTES;
        v16h a[4], b[4];
#pragma unroll
        for (int i = 0; i < 4; ++i) {
            char* pa = sa + i * 16 * ROW_STRIDE;
            a[i] = cat8(*(const v8h*)pa, *(const v8h*)(pa + 16));
            char* pb = sb + i * 16 * ROW_STRIDE;
            b[i] = cat8(*(const v8h*)pb, *(const v8h*)(pb + 16));
        }
#pragma unroll
        for (int i = 0; i < 4; ++i)
#pragma unroll
            for (int j = 0; j < 4; ++j)
                acc[i][j] = __builtin_amdgcn_wmma_f32_16x16x32_f16(
                    false, a[i], false, b[j], (short)0, acc[i][j], false, false);
    }

    // D layout: VGPR r -> M = r + 8*lhalf, N = lmod (within each 16x16 frag)
#pragma unroll
    for (int i = 0; i < 4; ++i) {
        const size_t mbase = (size_t)(m0 + wm * 64 + i * 16 + 8 * lhalf);
#pragma unroll
        for (int j = 0; j < 4; ++j) {
            const size_t nbase = (size_t)(n0 + wn * 64 + j * 16 + lmod);
#pragma unroll
            for (int r = 0; r < 8; ++r)
                out[(mbase + r) * (size_t)N_DIM + nbase] = acc[i][j][r];
        }
    }
}

// ---------------------------------------------------------------------------
extern "C" void kernel_launch(void* const* d_in, const int* in_sizes, int n_in,
                              void* d_out, int out_size, void* d_ws, size_t ws_size,
                              hipStream_t stream) {
    const float* x     = (const float*)d_in[0];   // [4,2048,4096]
    const float* w     = (const float*)d_in[1];   // [11008,4096]
    const float* scale = (const float*)d_in[2];   // [11008,1]
    float* out = (float*)d_out;                   // [4,2048,11008]

    // workspace layout (all offsets 256B aligned)
    char* ws = (char*)d_ws;
    const size_t WQ_BYTES = (size_t)N_DIM * K_DIM * 2;   // 90,177,536
    const size_t X_BYTES  = (size_t)M_DIM * K_DIM * 2;   // 67,108,864
    _Float16* wq16 = (_Float16*)ws;
    _Float16* x16  = (_Float16*)(ws + WQ_BYTES);
    float* partial = (float*)(ws + WQ_BYTES + X_BYTES);
    float* thresh  = partial + RED_BLOCKS;

    const int nw  = N_DIM * K_DIM;        // 45,088,768
    const int nw4 = nw / 4;
    const int nx4 = (M_DIM * K_DIM) / 4;  // 8,388,608

    // 1) threshold = 0.7 * mean(|W|)
    k_abs_partial<<<RED_BLOCKS, 256, 0, stream>>>(w, partial, nw4);
    k_thresh<<<1, 256, 0, stream>>>(partial, thresh, RED_BLOCKS, 1.0f / (float)nw);

    // 2) ternarize+scale weight -> f16
    k_quant<<<(nw4 + 255) / 256, 256, 0, stream>>>(w, scale, thresh, wq16, nw4);

    // 3) x -> f16
    k_xcvt<<<(nx4 + 255) / 256, 256, 0, stream>>>(x, x16, nx4);

    // 4) double-buffered async-LDS WMMA GEMM
    dim3 grid(N_DIM / BT_N, M_DIM / BT_M);  // (43, 32)
    k_gemm<<<grid, 512, 0, stream>>>(x16, wq16, out);
}
